// NeighborDecoder_21758304321717
// MI455X (gfx1250) — matrix operand order repaired and checked by reading, not compiled, not run on hardware
//
#include <hip/hip_runtime.h>

// ---------------------------------------------------------------------------
// NeighborDecoder for MI455X (gfx1250): bf16 WMMA GEMMs with 2x4 / 1x4
// register macro-tiles, fused nb softmax, L2-resident per-batch scan.
// Shapes: b=256, C=512, N=512 (8x64), T=50, 97 classes.
// ---------------------------------------------------------------------------

typedef __attribute__((ext_vector_type(16))) __bf16 v16bf;
typedef __attribute__((ext_vector_type(8)))  float  v8f;

#define B_   256
#define C_   512
#define N_   512
#define NP_  513   // N+1
#define MP_  544   // N+1 padded to 34 tiles of 16 (even # of tiles)
#define T_   50
#define NC_  97
#define PTH  0.6f
#define SCALE 0.044194173824159216f  // 1/sqrt(512)

union TileU {
    v16bf b;
    uint4 q[2];
    unsigned short s[16];
};

__device__ __forceinline__ unsigned short f2bf(float f) {
    unsigned u = __float_as_uint(f);
    unsigned r = u + 0x7FFFu + ((u >> 16) & 1u);  // round-to-nearest-even
    return (unsigned short)(r >> 16);
}
__device__ __forceinline__ float bf2f(unsigned short h) {
    return __uint_as_float(((unsigned)h) << 16);
}

// A fragment: 16x32 bf16, row-major source [rows][ld]; ISA 7.12.2 layout:
// lane L: row = L&15; VGPR0-3 hold K=kb..kb+7, VGPR4-7 hold K=kb+16..kb+23,
// kb = (L<16) ? 0 : 8 (plus 32*kt).
__device__ __forceinline__ v16bf load_a(const unsigned short* src, int ld,
                                        int mtile, int kt) {
    int lane = threadIdx.x & 31;
    int row  = mtile * 16 + (lane & 15);
    int kb   = kt * 32 + ((lane >> 4) << 3);  // 0 or 8
    TileU t;
    t.q[0] = *(const uint4*)(src + (long)row * ld + kb);
    t.q[1] = *(const uint4*)(src + (long)row * ld + kb + 16);
    return t.b;
}

// B fragment: 32x16 bf16; source row-major [Ncols][ld] holding B^T (i.e. the
// "weight rows"); lane L: col = L&15; K = kb..kb+7, kb+16..kb+23, kb = (L<16) ? 0 : 8.
__device__ __forceinline__ v16bf load_b(const unsigned short* src, int ld,
                                        int ntile, int kt) {
    int lane = threadIdx.x & 31;
    int col  = ntile * 16 + (lane & 15);
    int kb   = kt * 32 + ((lane >> 4) << 3);  // 0 or 8
    const uint4* p = (const uint4*)(src + (long)col * ld + kb);
    TileU t;
    t.q[0] = p[0];
    t.q[1] = p[2];
    return t.b;
}

__device__ __forceinline__ v8f wmma_bf16(v16bf a, v16bf b, v8f c) {
    return __builtin_amdgcn_wmma_f32_16x16x32_bf16(
        false, a, false, b, (short)0, c, false, false);
}

// C/D layout store: VGPR i -> row rbase+i, col = ntile*16 + (lane&15)
__device__ __forceinline__ void store_tile(unsigned short* Db, int ldd,
                                           int mtile, int ntile, v8f acc,
                                           const float* __restrict__ bias) {
    int lane  = threadIdx.x & 31;
    int col   = ntile * 16 + (lane & 15);
    int rbase = mtile * 16 + ((lane >> 4) ? 8 : 0);
    float bi  = bias[col];
    for (int i = 0; i < 8; ++i)
        Db[(long)(rbase + i) * ldd + col] = f2bf(acc[i] + bi);
}

// --------------------------- prep kernels ----------------------------------

__global__ void cvt_bf16_kernel(const float* __restrict__ src,
                                unsigned short* __restrict__ dst, int n) {
    int i = blockIdx.x * 256 + threadIdx.x;
    if (i < n) dst[i] = f2bf(src[i]);
}

// xt[b][n][c] = bf16(x[b][c][n]); 16x16 LDS tiled transpose. grid:(1024,B)
__global__ void transpose_kernel(const float* __restrict__ x,
                                 unsigned short* __restrict__ xt) {
    __shared__ float tile[16][17];
    int b  = blockIdx.y;
    int ct = blockIdx.x & 31;   // c tile
    int nt = blockIdx.x >> 5;   // n tile
    int lx = threadIdx.x & 15;
    int ly = threadIdx.x >> 4;
    tile[ly][lx] = x[((long)b * C_ + ct * 16 + ly) * (long)N_ + nt * 16 + lx];
    __syncthreads();
    xt[((long)b * MP_ + nt * 16 + ly) * (long)C_ + ct * 16 + lx] =
        f2bf(tile[lx][ly]);
}

// row N_ = eos embedding; rows N_+1..MP_-1 = 0. grid: B
__global__ void pad_kernel(const float* __restrict__ eos,
                           unsigned short* __restrict__ xt) {
    int b = blockIdx.x, tid = threadIdx.x;
    unsigned short* base = xt + (long)b * MP_ * C_;
    for (int c = tid; c < C_; c += 256) base[(long)N_ * C_ + c] = f2bf(eos[c]);
    for (long i = tid; i < (long)(MP_ - NP_) * C_; i += 256)
        base[(long)NP_ * C_ + i] = 0;
}

// g[b][c] = masked mean over n. grid: B
__global__ void pool_kernel(const float* __restrict__ x,
                            const float* __restrict__ mask,
                            float* __restrict__ g) {
    __shared__ float red[256];
    __shared__ float minv;
    int b = blockIdx.x, tid = threadIdx.x;
    const float* mrow = mask + (long)b * N_;
    float pm = 0.f;
    for (int n = tid; n < N_; n += 256) pm += mrow[n];
    red[tid] = pm;
    __syncthreads();
    for (int s = 128; s; s >>= 1) {
        if (tid < s) red[tid] += red[tid + s];
        __syncthreads();
    }
    if (tid == 0) minv = 1.f / (red[0] + 1e-10f);
    __syncthreads();
    for (int c = tid; c < C_; c += 256) {
        const float* row = x + ((long)b * C_ + c) * (long)N_;
        float s = 0.f;
        for (int n = 0; n < N_; ++n) s += row[n] * mrow[n];
        g[(long)b * C_ + c] = s * minv;
    }
}

// q_start[b] = g[b] @ qW^T + qb. grid: B
__global__ void qstart_kernel(const float* __restrict__ g,
                              const float* __restrict__ qW,
                              const float* __restrict__ qb,
                              float* __restrict__ qs) {
    __shared__ float gs[C_];
    int b = blockIdx.x, tid = threadIdx.x;
    gs[tid]       = g[(long)b * C_ + tid];
    gs[tid + 256] = g[(long)b * C_ + tid + 256];
    __syncthreads();
    for (int c = tid; c < C_; c += 256) {
        const float* w = qW + (long)c * C_;
        float s = 0.f;
        for (int e = 0; e < C_; ++e) s += w[e] * gs[e];
        qs[(long)b * C_ + c] = s + qb[c];
    }
}

// Fold: Wq2t[d][e] = sum_c qW[c][e]*bw[c][d]; bias2[d] = sum_c qb[c]*bw[c][d].
// grid: C_ (one block per d)
__global__ void wq2_kernel(const float* __restrict__ qW,
                           const float* __restrict__ bw,
                           const float* __restrict__ qb,
                           unsigned short* __restrict__ wq2t,
                           float* __restrict__ bias2) {
    int d = blockIdx.x, tid = threadIdx.x;
    for (int e = tid; e < C_; e += 256) {
        float s = 0.f;
        for (int c = 0; c < C_; ++c)
            s += qW[(long)c * C_ + e] * bw[(long)c * C_ + d];
        wq2t[(long)d * C_ + e] = f2bf(s);
    }
    if (tid == 0) {
        float s = 0.f;
        for (int c = 0; c < C_; ++c) s += qb[c] * bw[(long)c * C_ + d];
        bias2[d] = s;
    }
}

// --------------------------- WMMA GEMMs ------------------------------------

// D[b] = A[b] @ B^T + bias. 2 M-tiles x 4 N-tiles per wave (8 accumulators);
// all 8 waves in a block share the same M-pair (A rows hit L0).
// grid: (mpairs, B), blockDim 256.
__global__ void gemm_bf16_kernel(const unsigned short* __restrict__ A,
                                 const unsigned short* __restrict__ Bw,
                                 const float* __restrict__ bias,
                                 unsigned short* __restrict__ D,
                                 long astride, long dstride) {
    int b      = blockIdx.y;
    int mtile0 = blockIdx.x * 2;
    int mtile1 = mtile0 + 1;
    int n0     = (threadIdx.x >> 5) * 4;  // wave id * 4 n-tiles
    const unsigned short* Ab = A + (long)b * astride;
    v8f a00 = {}, a01 = {}, a02 = {}, a03 = {};
    v8f a10 = {}, a11 = {}, a12 = {}, a13 = {};
    for (int kt = 0; kt < 16; ++kt) {
        v16bf fa0 = load_a(Ab, C_, mtile0, kt);
        v16bf fa1 = load_a(Ab, C_, mtile1, kt);
        v16bf fb0 = load_b(Bw, C_, n0 + 0, kt);
        v16bf fb1 = load_b(Bw, C_, n0 + 1, kt);
        v16bf fb2 = load_b(Bw, C_, n0 + 2, kt);
        v16bf fb3 = load_b(Bw, C_, n0 + 3, kt);
        a00 = wmma_bf16(fa0, fb0, a00);
        a01 = wmma_bf16(fa0, fb1, a01);
        a02 = wmma_bf16(fa0, fb2, a02);
        a03 = wmma_bf16(fa0, fb3, a03);
        a10 = wmma_bf16(fa1, fb0, a10);
        a11 = wmma_bf16(fa1, fb1, a11);
        a12 = wmma_bf16(fa1, fb2, a12);
        a13 = wmma_bf16(fa1, fb3, a13);
    }
    unsigned short* Db = D + (long)b * dstride;
    store_tile(Db, C_, mtile0, n0 + 0, a00, bias);
    store_tile(Db, C_, mtile0, n0 + 1, a01, bias);
    store_tile(Db, C_, mtile0, n0 + 2, a02, bias);
    store_tile(Db, C_, mtile0, n0 + 3, a03, bias);
    store_tile(Db, C_, mtile1, n0 + 0, a10, bias);
    store_tile(Db, C_, mtile1, n0 + 1, a11, bias);
    store_tile(Db, C_, mtile1, n0 + 2, a12, bias);
    store_tile(Db, C_, mtile1, n0 + 3, a13, bias);
}

// nb_map rows [itile*16 .. +15]: logits = (t @ kf^T + bb)*scale, mask,
// row-softmax, write f32 nb_map. Wave w owns j-tiles {w, w+8, w+16, w+24}
// sharing one A fragment per K step; wave 0 mops up j-tile 32.
// grid: (32 itiles, B), blockDim 256.
__global__ void nb_kernel(const unsigned short* __restrict__ Tm,
                          const unsigned short* __restrict__ Kf,
                          const float* __restrict__ mask,
                          const float* __restrict__ bbp,
                          float* __restrict__ nb_out) {
    __shared__ float lds[16][544];
    int b = blockIdx.y, itile = blockIdx.x;
    int wave = threadIdx.x >> 5, lane = threadIdx.x & 31;
    const unsigned short* Tb = Tm + (long)b * N_ * C_;
    const unsigned short* Kb = Kf + (long)b * MP_ * C_;
    float bb = bbp[0];
    {
        v8f a0 = {}, a1 = {}, a2 = {}, a3 = {};
        for (int kt = 0; kt < 16; ++kt) {
            v16bf fa = load_a(Tb, C_, itile, kt);
            a0 = wmma_bf16(fa, load_b(Kb, C_, wave + 0,  kt), a0);
            a1 = wmma_bf16(fa, load_b(Kb, C_, wave + 8,  kt), a1);
            a2 = wmma_bf16(fa, load_b(Kb, C_, wave + 16, kt), a2);
            a3 = wmma_bf16(fa, load_b(Kb, C_, wave + 24, kt), a3);
        }
        int rbase = (lane >> 4) ? 8 : 0;
        int c0 = (wave + 0)  * 16 + (lane & 15);
        int c1 = (wave + 8)  * 16 + (lane & 15);
        int c2 = (wave + 16) * 16 + (lane & 15);
        int c3 = (wave + 24) * 16 + (lane & 15);
        for (int i = 0; i < 8; ++i) {
            lds[rbase + i][c0] = (a0[i] + bb) * SCALE;
            lds[rbase + i][c1] = (a1[i] + bb) * SCALE;
            lds[rbase + i][c2] = (a2[i] + bb) * SCALE;
            lds[rbase + i][c3] = (a3[i] + bb) * SCALE;
        }
    }
    if (wave == 0) {  // wave-uniform: j-tile 32 (column 512 + pad)
        v8f a0 = {};
        for (int kt = 0; kt < 16; ++kt)
            a0 = wmma_bf16(load_a(Tb, C_, itile, kt),
                           load_b(Kb, C_, 32, kt), a0);
        int col   = 32 * 16 + (lane & 15);
        int rbase = (lane >> 4) ? 8 : 0;
        if (col < NP_)
            for (int i = 0; i < 8; ++i)
                lds[rbase + i][col] = (a0[i] + bb) * SCALE;
    }
    __syncthreads();
    // 16 threads per row softmax
    int r = threadIdx.x >> 4, sub = threadIdx.x & 15;
    const float* mrow = mask + (long)b * N_;
    float mx = -INFINITY;
    for (int j = sub; j < NP_; j += 16) {
        float v = lds[r][j];
        if (j < N_ && mrow[j] < 0.5f) { v = -INFINITY; lds[r][j] = v; }
        mx = fmaxf(mx, v);
    }
    for (int o = 8; o; o >>= 1) mx = fmaxf(mx, __shfl_xor(mx, o, 16));
    float sum = 0.f;
    for (int j = sub; j < NP_; j += 16) sum += __expf(lds[r][j] - mx);
    for (int o = 8; o; o >>= 1) sum += __shfl_xor(sum, o, 16);
    float inv = 1.f / sum;
    float* orow = nb_out + (long)b * NP_ * NP_ + (long)(itile * 16 + r) * NP_;
    for (int j = sub; j < NP_; j += 16) orow[j] = __expf(lds[r][j] - mx) * inv;
}

// start logits + softmax -> start_map. grid: B
__global__ void start_kernel(const unsigned short* __restrict__ Kf,
                             const float* __restrict__ qstart,
                             const float* __restrict__ mask,
                             float* __restrict__ smap) {
    __shared__ float q[C_];
    __shared__ float logit[NP_];
    __shared__ float red[256];
    int b = blockIdx.x, tid = threadIdx.x;
    q[tid]       = qstart[(long)b * C_ + tid];
    q[tid + 256] = qstart[(long)b * C_ + tid + 256];
    __syncthreads();
    const unsigned short* Kb = Kf + (long)b * MP_ * C_;
    for (int j = tid; j < NP_; j += 256) {
        float s = 0.f;
        const unsigned short* row = Kb + (long)j * C_;
        for (int c = 0; c < C_; ++c) s += bf2f(row[c]) * q[c];
        float v = s * SCALE;
        if (j < N_ && mask[(long)b * N_ + j] < 0.5f) v = -INFINITY;
        logit[j] = v;
    }
    __syncthreads();
    float mx = -INFINITY;
    for (int j = tid; j < NP_; j += 256) mx = fmaxf(mx, logit[j]);
    red[tid] = mx; __syncthreads();
    for (int s = 128; s; s >>= 1) {
        if (tid < s) red[tid] = fmaxf(red[tid], red[tid + s]);
        __syncthreads();
    }
    mx = red[0]; __syncthreads();
    float sum = 0.f;
    for (int j = tid; j < NP_; j += 256) sum += __expf(logit[j] - mx);
    red[tid] = sum; __syncthreads();
    for (int s = 128; s; s >>= 1) {
        if (tid < s) red[tid] += red[tid + s];
        __syncthreads();
    }
    float inv = 1.f / red[0];
    for (int j = tid; j < NP_; j += 256)
        smap[(long)b * NP_ + j] = __expf(logit[j] - mx) * inv;
}

// Per-batch scan: nb_map[b] (1.05 MB) stays L2-resident over all 50 steps.
// grid: B, blockDim 256.
__global__ void scan_kernel(const float* __restrict__ smap,
                            const unsigned short* __restrict__ Xt,
                            float* __restrict__ nb,
                            float* __restrict__ cm_out,
                            float* __restrict__ msk_out,
                            float* __restrict__ cf_out) {
    __shared__ float cm[NP_];
    __shared__ float ncm[NP_];
    int b = blockIdx.x, tid = threadIdx.x;
    float* nbb = nb + (long)b * NP_ * NP_;
    // appended one-hot last row of nb_map
    for (int j = tid; j < NP_; j += 256)
        nbb[(long)N_ * NP_ + j] = (j == N_) ? 1.f : 0.f;
    for (int j = tid; j < NP_; j += 256) cm[j] = smap[(long)b * NP_ + j];
    __syncthreads();
    int fin = 0;
    const unsigned short* Xb = Xt + (long)b * MP_ * C_;
    for (int t = 0; t < T_; ++t) {
        // emit char_map + mask
        for (int j = tid; j < NP_; j += 256)
            cm_out[((long)b * T_ + t) * NP_ + j] = cm[j];
        float mk = (fin == 0) ? 1.f : 0.f;
        if (tid == 0) msk_out[(long)b * T_ + t] = mk;
        // char feats (coalesced over c; cm[n] broadcast from LDS)
        float f0 = 0.f, f1 = 0.f;
        for (int n = 0; n < NP_; ++n) {
            float w = cm[n];
            f0 += w * bf2f(Xb[(long)n * C_ + tid]);
            f1 += w * bf2f(Xb[(long)n * C_ + tid + 256]);
        }
        cf_out[((long)b * T_ + t) * C_ + tid]       = f0 * mk;
        cf_out[((long)b * T_ + t) * C_ + tid + 256] = f1 * mk;
        // EOS accumulation then advance
        if (cm[N_] > PTH) fin++;
        for (int j = tid; j < NP_; j += 256) {
            float a = 0.f;
            for (int n = 0; n < NP_; ++n) a += cm[n] * nbb[(long)n * NP_ + j];
            ncm[j] = a;
        }
        __syncthreads();
        for (int j = tid; j < NP_; j += 256) cm[j] = ncm[j];
        __syncthreads();
    }
}

// logits = char_feats @ clsW^T + clsb. grid: B
__global__ void logits_kernel(const float* __restrict__ cf,
                              const float* __restrict__ clsW,
                              const float* __restrict__ clsb,
                              float* __restrict__ out) {
    int b = blockIdx.x, tid = threadIdx.x;
    const float* cfb = cf + (long)b * T_ * C_;
    for (int idx = tid; idx < T_ * NC_; idx += 256) {
        int t = idx / NC_, cls = idx % NC_;
        const float* f = cfb + (long)t * C_;
        const float* w = clsW + (long)cls * C_;
        float s = 0.f;
        for (int c = 0; c < C_; ++c) s += f[c] * w[c];
        out[((long)b * T_ + t) * NC_ + cls] = s + clsb[cls];
    }
}

// --------------------------- launcher --------------------------------------

extern "C" void kernel_launch(void* const* d_in, const int* in_sizes, int n_in,
                              void* d_out, int out_size, void* d_ws,
                              size_t ws_size, hipStream_t stream) {
    const float* x    = (const float*)d_in[0];
    const float* mask = (const float*)d_in[1];
    const float* eos  = (const float*)d_in[2];
    const float* qW   = (const float*)d_in[3];
    const float* qb   = (const float*)d_in[4];
    const float* kW   = (const float*)d_in[5];
    const float* kb   = (const float*)d_in[6];
    const float* bw   = (const float*)d_in[7];  // [1,C,C]
    const float* bb   = (const float*)d_in[8];
    const float* clsW = (const float*)d_in[9];
    const float* clsb = (const float*)d_in[10];
    (void)in_sizes; (void)n_in; (void)out_size; (void)ws_size;

    // workspace carve-up
    char* w = (char*)d_ws;
    unsigned short* xt   = (unsigned short*)w;  w += (long)B_ * MP_ * C_ * 2;
    unsigned short* kf   = (unsigned short*)w;  w += (long)B_ * MP_ * C_ * 2;
    unsigned short* tm   = (unsigned short*)w;  w += (long)B_ * N_  * C_ * 2;
    unsigned short* kwb  = (unsigned short*)w;  w += (long)C_ * C_ * 2;
    unsigned short* wq2t = (unsigned short*)w;  w += (long)C_ * C_ * 2;
    float* bias2 = (float*)w;  w += (long)C_ * 4;
    float* g     = (float*)w;  w += (long)B_ * C_ * 4;
    float* qs    = (float*)w;  w += (long)B_ * C_ * 4;
    float* smap  = (float*)w;  w += (long)B_ * NP_ * 4;
    float* cf    = (float*)w;  w += (long)B_ * T_ * C_ * 4;

    // output carve-up (flat, return order)
    float* out_logits = (float*)d_out;                         // [B,T,97]
    float* out_cm     = out_logits + (long)B_ * T_ * NC_;      // [B,T,513]
    float* out_msk    = out_cm + (long)B_ * T_ * NP_;          // [B,T]
    float* out_nb     = out_msk + (long)B_ * T_;               // [B,513,513]

    // 1. weight + activation conversions
    cvt_bf16_kernel<<<(C_ * C_ + 255) / 256, 256, 0, stream>>>(kW, kwb, C_ * C_);
    transpose_kernel<<<dim3(1024, B_), 256, 0, stream>>>(x, xt);
    pad_kernel<<<B_, 256, 0, stream>>>(eos, xt);
    // 2. pooled query + folded bilinear weight
    pool_kernel<<<B_, 256, 0, stream>>>(x, mask, g);
    qstart_kernel<<<B_, 256, 0, stream>>>(g, qW, qb, qs);
    wq2_kernel<<<C_, 256, 0, stream>>>(qW, bw, qb, wq2t, bias2);
    // 3. WMMA GEMMs: k_feat [B,544,512] (34 mtiles -> 17 pairs),
    //    t = xf@Wq2 [B,512,512] (32 mtiles -> 16 pairs)
    gemm_bf16_kernel<<<dim3(17, B_), 256, 0, stream>>>(
        xt, kwb, kb, kf, (long)MP_ * C_, (long)MP_ * C_);
    gemm_bf16_kernel<<<dim3(16, B_), 256, 0, stream>>>(
        xt, wq2t, bias2, tm, (long)MP_ * C_, (long)N_ * C_);
    // 4. start map + fused nb GEMM/softmax
    start_kernel<<<B_, 256, 0, stream>>>(kf, qs, mask, smap);
    nb_kernel<<<dim3(32, B_), 256, 0, stream>>>(tm, kf, mask, bb, out_nb);
    // 5. scan recurrence + char feats, then classifier head
    scan_kernel<<<B_, 256, 0, stream>>>(smap, xt, out_nb, out_cm, out_msk, cf);
    logits_kernel<<<B_, 256, 0, stream>>>(cf, clsW, clsb, out_logits);
}
